// Mix9Net_16295105921530
// MI455X (gfx1250) — compile-verified, matrix operand order checked by reading, not run on hardware
//
#include <hip/hip_runtime.h>
#include <hip/hip_bf16.h>

// Mix9Net directional-conv tower for MI455X (gfx1250), wave32 + WMMA f16.
// One workgroup = one (batch, direction) pair; all activations resident in LDS.

typedef __attribute__((ext_vector_type(16))) _Float16 v16h;
typedef __attribute__((ext_vector_type(8)))  _Float16 v8h;
typedef __attribute__((ext_vector_type(8)))  float    v8f;

// f16 weight workspace layout (element = half)
#define WS_RBDW 0            // rb_dconv_w : 4*3*128*128 = 196608
#define WS_RBPW 196608       // rb_c1_w    : 4*128*128   = 65536
#define WS_C0W1 262144       // c0_w1      : 128*128     = 16384
#define WS_C0W2 278528       // c0_w2      : 128*128     = 16384
#define WS_FINW 294912       // final_w    : 64*128      = 8192
#define WS_TOTAL 303104

#define P_HALVES (289*128)   // 17x17 zero-haloed activation buffer
#define T_HALVES (240*128)   // 15 tiles * 16 px staging buffer

__device__ __forceinline__ float silu_f(float x) {
    return x * (1.0f / (1.0f + __expf(-x)));
}

// Load one 16-half WMMA fragment: halves [0..7] = K kb..kb+7, [8..15] = K kb+16..kb+23.
// Two 16-byte loads (ds_load_b128 from LDS / global_load_b128 from global).
__device__ __forceinline__ v16h load_frag(const _Float16* p) {
    union { v16h v; struct { v8h lo, hi; } s; } u;
    u.s.lo = *(const v8h*)p;
    u.s.hi = *(const v8h*)(p + 16);
    return u.v;
}

__device__ __forceinline__ v8f wmma_f16(v16h a, v16h b, v8f c) {
    return __builtin_amdgcn_wmma_f32_16x16x32_f16(
        /*neg_a=*/false, a, /*neg_b=*/false, b,
        /*c_mod=*/(short)0, c, /*reuse_a=*/false, /*reuse_b=*/false);
}

// Preload 4 A-fragments (K=128) for one 16-row tile of a row-major [O][128] weight.
__device__ __forceinline__ void load_a4(v16h* a, const _Float16* w, int row, int kb) {
#pragma unroll
    for (int ks = 0; ks < 4; ++ks)
        a[ks] = load_frag(w + row * 128 + ks * 32 + kb);
}

// acc += W(16x128) * src(128x16); src points at pixel-major [px][128] halves for this tile.
__device__ __forceinline__ v8f mm_k128(v8f acc, const v16h* a, const _Float16* src, int kb) {
#pragma unroll
    for (int ks = 0; ks < 4; ++ks) {
        v16h b = load_frag(src + ks * 32 + kb);
        acc = wmma_f16(a[ks], b, acc);
    }
    return acc;
}

__global__ void cvt_weights(const float* rbdw, const float* rbpw, const float* w1,
                            const float* w2, const float* fw, _Float16* outw) {
    for (int i = blockIdx.x * blockDim.x + threadIdx.x; i < WS_TOTAL;
         i += gridDim.x * blockDim.x) {
        int j = i;
        float v;
        if (j < 196608)               v = rbdw[j];
        else if ((j -= 196608) < 65536) v = rbpw[j];
        else if ((j -= 65536)  < 16384) v = w1[j];
        else if ((j -= 16384)  < 16384) v = w2[j];
        else                            v = fw[j - 16384];
        outw[i] = (_Float16)v;
    }
}

__global__ __launch_bounds__(256, 1)
void mix9_main(const float* __restrict__ x,
               const float* __restrict__ dconv0_w, const float* __restrict__ dconv0_b,
               const float* __restrict__ rb_dconv_b, const float* __restrict__ rb_c1_b,
               const float* __restrict__ c0_b1, const float* __restrict__ c0_b2,
               const float* __restrict__ final_b,
               const _Float16* __restrict__ wgt,
               float* __restrict__ out) {
    extern __shared__ char smem[];
    _Float16* P0 = (_Float16*)smem;
    _Float16* P1 = P0 + P_HALVES;
    _Float16* T1 = P1 + P_HALVES;
    float*    XS = (float*)(T1 + T_HALVES);

    const int bd   = blockIdx.x;          // b*4 + d
    const int b    = bd >> 2;
    const int d    = bd & 3;
    const int tid  = threadIdx.x;
    const int lane = tid & 31;
    const int wave = tid >> 5;
    const int col  = lane & 15;
    const int kb   = (lane >> 4) * 8;     // K sub-offset per WMMA 16-bit layout

    // constant linear shifts in the 17-wide haloed buffer per direction
    int offA, offB;
    switch (d) {
        case 0:  offA = -1;  offB = 1;   break;   // horizontal
        case 1:  offA = -17; offB = 17;  break;   // vertical
        case 2:  offA = -18; offB = 18;  break;   // diagonal
        default: offA = 16;  offB = -16; break;   // anti-diagonal
    }

    // ---- zero both haloed buffers + staging halo of XS ----
    {
        uint4* z = (uint4*)smem;
        const int nz = (2 * P_HALVES * 2) / 16;   // bytes/16
        for (int i = tid; i < nz; i += 256) z[i] = make_uint4(0u, 0u, 0u, 0u);
        for (int i = tid; i < 289 * 2; i += 256) XS[i] = 0.0f;
    }
    __syncthreads();
    // stage input image [2][15][15] into haloed XS [289][2]
    for (int i = tid; i < 450; i += 256) {
        int ci = i / 225, t = i % 225;
        int p = (t / 15 + 1) * 17 + (t % 15) + 1;
        XS[p * 2 + ci] = x[((size_t)b * 2 + ci) * 225 + t];
    }
    __syncthreads();

    // ---- layer 0: directional conv 2->128 + SiLU (VALU, tiny K) -> P0 interior ----
    for (int i = tid; i < 225 * 128; i += 256) {
        int oc = i & 127, t = i >> 7;
        int p = (t / 15 + 1) * 17 + (t % 15) + 1;
        int pa = p + offA, pb = p + offB;
        float s = dconv0_b[oc];
        s += dconv0_w[0 * 256 + oc * 2 + 0] * XS[pa * 2 + 0]
           + dconv0_w[0 * 256 + oc * 2 + 1] * XS[pa * 2 + 1];
        s += dconv0_w[1 * 256 + oc * 2 + 0] * XS[p * 2 + 0]
           + dconv0_w[1 * 256 + oc * 2 + 1] * XS[p * 2 + 1];
        s += dconv0_w[2 * 256 + oc * 2 + 0] * XS[pb * 2 + 0]
           + dconv0_w[2 * 256 + oc * 2 + 1] * XS[pb * 2 + 1];
        P0[p * 128 + oc] = (_Float16)silu_f(s);
    }
    __syncthreads();

    _Float16* Pin  = P0;
    _Float16* Pout = P1;
    const int ot   = wave;            // each wave owns one 16-channel output tile
    const int chb  = ot * 16 + kb;    // C/D channel base for this lane
    const int row  = ot * 16 + col;   // A-matrix row for this lane

    // ---- 4 directional res blocks ----
    for (int ib = 0; ib < 4; ++ib) {
        // directional conv 128->128 (3 shifted GEMMs), SiLU -> T1
        v16h a[12];
        {
            const _Float16* w = wgt + WS_RBDW + ib * 3 * 16384;
            load_a4(a + 0, w,          row, kb);
            load_a4(a + 4, w + 16384,  row, kb);
            load_a4(a + 8, w + 32768,  row, kb);
        }
        for (int t = 0; t < 15; ++t) {
            int tp = t * 16 + col;
            int p = (tp < 225) ? ((tp / 15 + 1) * 17 + (tp % 15) + 1) : 18;
            v8f acc = {0.f, 0.f, 0.f, 0.f, 0.f, 0.f, 0.f, 0.f};
            acc = mm_k128(acc, a + 0, Pin + (p + offA) * 128, kb);
            acc = mm_k128(acc, a + 4, Pin + p * 128,          kb);
            acc = mm_k128(acc, a + 8, Pin + (p + offB) * 128, kb);
            v8f bb = *(const v8f*)(rb_dconv_b + ib * 128 + chb);
            v8h o;
#pragma unroll
            for (int v = 0; v < 8; ++v) o[v] = (_Float16)silu_f(acc[v] + bb[v]);
            *(v8h*)(T1 + tp * 128 + chb) = o;
        }
        __syncthreads();

        // pointwise 128->128, SiLU, + residual -> Pout
        v16h aw[4];
        load_a4(aw, wgt + WS_RBPW + ib * 16384, row, kb);
        for (int t = 0; t < 15; ++t) {
            int tp = t * 16 + col;
            bool valid = tp < 225;
            int p = valid ? ((tp / 15 + 1) * 17 + (tp % 15) + 1) : 18;
            v8f acc = {0.f, 0.f, 0.f, 0.f, 0.f, 0.f, 0.f, 0.f};
            acc = mm_k128(acc, aw, T1 + tp * 128, kb);
            v8f bb = *(const v8f*)(rb_c1_b + ib * 128 + chb);
            v8h r  = *(const v8h*)(Pin + p * 128 + chb);
            v8h o;
#pragma unroll
            for (int v = 0; v < 8; ++v)
                o[v] = (_Float16)(silu_f(acc[v] + bb[v]) + (float)r[v]);
            if (valid) *(v8h*)(Pout + p * 128 + chb) = o;
        }
        __syncthreads();
        _Float16* tmp = Pin; Pin = Pout; Pout = tmp;
    }

    // ---- Conv0d res block: two pointwise 128->128 ----
    {
        v16h aw[4];
        load_a4(aw, wgt + WS_C0W1, row, kb);
        for (int t = 0; t < 15; ++t) {
            int tp = t * 16 + col;
            int p = (tp < 225) ? ((tp / 15 + 1) * 17 + (tp % 15) + 1) : 18;
            v8f acc = {0.f, 0.f, 0.f, 0.f, 0.f, 0.f, 0.f, 0.f};
            acc = mm_k128(acc, aw, Pin + p * 128, kb);
            v8f bb = *(const v8f*)(c0_b1 + chb);
            v8h o;
#pragma unroll
            for (int v = 0; v < 8; ++v) o[v] = (_Float16)silu_f(acc[v] + bb[v]);
            *(v8h*)(T1 + tp * 128 + chb) = o;
        }
        __syncthreads();

        load_a4(aw, wgt + WS_C0W2, row, kb);
        for (int t = 0; t < 15; ++t) {
            int tp = t * 16 + col;
            bool valid = tp < 225;
            int p = valid ? ((tp / 15 + 1) * 17 + (tp % 15) + 1) : 18;
            v8f acc = {0.f, 0.f, 0.f, 0.f, 0.f, 0.f, 0.f, 0.f};
            acc = mm_k128(acc, aw, T1 + tp * 128, kb);
            v8f bb = *(const v8f*)(c0_b2 + chb);
            v8h r  = *(const v8h*)(Pin + p * 128 + chb);
            v8h o;
#pragma unroll
            for (int v = 0; v < 8; ++v)
                o[v] = (_Float16)(silu_f(acc[v] + bb[v]) + (float)r[v]);
            if (valid) *(v8h*)(Pout + p * 128 + chb) = o;
        }
        __syncthreads();
        _Float16* tmp = Pin; Pin = Pout; Pout = tmp;
    }

    // ---- final pointwise 128->64 (no SiLU) -> global out [B][4][64][15][15] ----
    {
        const int fot  = wave & 3;             // 4 output-channel tiles
        const int frow = fot * 16 + col;
        const int fchb = fot * 16 + kb;
        v16h aw[4];
        load_a4(aw, wgt + WS_FINW, frow, kb);
        v8f bb = *(const v8f*)(final_b + fchb);
        const int ts = (wave >> 2) ? 8 : 0;
        const int te = (wave >> 2) ? 15 : 8;
        float* outbase = out + ((size_t)bd * 64) * 225;
        for (int t = ts; t < te; ++t) {
            int tp = t * 16 + col;
            bool valid = tp < 225;
            int p = valid ? ((tp / 15 + 1) * 17 + (tp % 15) + 1) : 18;
            v8f acc = {0.f, 0.f, 0.f, 0.f, 0.f, 0.f, 0.f, 0.f};
            acc = mm_k128(acc, aw, Pin + p * 128, kb);
            if (valid) {
#pragma unroll
                for (int v = 0; v < 8; ++v)
                    outbase[(size_t)(fchb + v) * 225 + tp] = acc[v] + bb[v];
            }
        }
    }
}

extern "C" void kernel_launch(void* const* d_in, const int* in_sizes, int n_in,
                              void* d_out, int out_size, void* d_ws, size_t ws_size,
                              hipStream_t stream) {
    const float* x        = (const float*)d_in[0];
    const float* d0w      = (const float*)d_in[1];
    const float* d0b      = (const float*)d_in[2];
    const float* rbdw     = (const float*)d_in[3];
    const float* rbdb     = (const float*)d_in[4];
    const float* rbpw     = (const float*)d_in[5];
    const float* rbpb     = (const float*)d_in[6];
    const float* c0w1     = (const float*)d_in[7];
    const float* c0b1     = (const float*)d_in[8];
    const float* c0w2     = (const float*)d_in[9];
    const float* c0b2     = (const float*)d_in[10];
    const float* finw     = (const float*)d_in[11];
    const float* finb     = (const float*)d_in[12];
    _Float16*    wgt      = (_Float16*)d_ws;
    float*       outp     = (float*)d_out;

    cvt_weights<<<128, 256, 0, stream>>>(rbdw, rbpw, c0w1, c0w2, finw, wgt);

    const size_t shmem = (size_t)(2 * P_HALVES + T_HALVES) * sizeof(_Float16)
                       + (size_t)289 * 2 * sizeof(float);   // 211,720 bytes
    mix9_main<<<256 * 4, 256, shmem, stream>>>(x, d0w, d0b, rbdb, rbpb,
                                               c0b1, c0b2, finb, wgt, outp);
}
